// EfficientSelfAttention2d_78417512890794
// MI455X (gfx1250) — matrix operand order repaired and checked by reading, not compiled, not run on hardware
//
#include <hip/hip_runtime.h>

// B=4, C=256, H=W=256, N=65536, rc=32
#define Bb 4
#define Cc 256
#define RC 32
#define Nn 65536

typedef __bf16           v16bf __attribute__((ext_vector_type(16)));
typedef __bf16           v2bf  __attribute__((ext_vector_type(2)));
typedef float            v8f   __attribute__((ext_vector_type(8)));
typedef unsigned         v8ui  __attribute__((ext_vector_type(8)));
typedef unsigned         v4u   __attribute__((ext_vector_type(4)));
typedef int              v4i   __attribute__((ext_vector_type(4)));
typedef int              v8i   __attribute__((ext_vector_type(8)));

#define GAS __attribute__((address_space(1)))
#define AS_G(T, p) ((T GAS*)(uintptr_t)(p))

#if __has_builtin(__builtin_amdgcn_tensor_load_to_lds)
#define HAVE_TDM 1
#warning "CDNA5 probe: tensor_load_to_lds builtin AVAILABLE"
#else
#define HAVE_TDM 0
#warning "CDNA5 probe: tensor_load_to_lds builtin NOT available"
#endif
#if __has_builtin(__builtin_amdgcn_cvt_pk_bf16_f32)
#warning "CDNA5 probe: cvt_pk_bf16_f32 builtin AVAILABLE"
#else
#warning "CDNA5 probe: cvt_pk_bf16_f32 builtin NOT available"
#endif

// fp32 -> bf16, round-half-up (cheap: 1 add + shift)
__device__ __forceinline__ unsigned short f2bf(float f) {
    unsigned u = __builtin_bit_cast(unsigned, f) + 0x8000u;
    return (unsigned short)(u >> 16);
}
__device__ __forceinline__ float bf2f(unsigned short h) {
    unsigned u = ((unsigned)h) << 16;
    return __builtin_bit_cast(float, u);
}
// pack two fp32 -> two bf16 in one dword.
// hw cvt if present, else round-half-up + one v_perm_b32 (3 VALU total)
__device__ __forceinline__ unsigned pk2(float a, float b) {
#if __has_builtin(__builtin_amdgcn_cvt_pk_bf16_f32)
    v2bf p = __builtin_amdgcn_cvt_pk_bf16_f32(a, b);
    return __builtin_bit_cast(unsigned, p);
#else
    unsigned ua = __builtin_bit_cast(unsigned, a) + 0x8000u;
    unsigned ub = __builtin_bit_cast(unsigned, b) + 0x8000u;
    // result[15:0]=ua[31:16], result[31:16]=ub[31:16]
    return __builtin_amdgcn_perm(ub, ua, 0x07060302u);
#endif
}
__device__ __forceinline__ v8f wmma_bf16(v8ui a, v8ui b, v8f c) {
    return __builtin_amdgcn_wmma_f32_16x16x32_bf16(
        false, __builtin_bit_cast(v16bf, a),
        false, __builtin_bit_cast(v16bf, b),
        (short)0, c, false, false);
}
__device__ __forceinline__ v8f vzero8() {
    v8f z = {0.f, 0.f, 0.f, 0.f, 0.f, 0.f, 0.f, 0.f};
    return z;
}

// ---------------- Kernel 0: zero context accumulator ----------------
__global__ void kZero(float* p, int n) {
    int i = blockIdx.x * 256 + threadIdx.x;
    if (i < n) p[i] = 0.f;
}

// ---------------- Kernel 0b: pre-pack weights into A-fragment order ----
// wbuf layout: dword[((t*8 + kk)*32 + lane)*8 + j]
//   t: 16-row output tile (0,1=q ; 2,3=k ; 4..19=v), kk: K-step (32 ch)
// A-frag (16-bit A 16x32), this code's k order (B matches): lane m=lane&15, half=lane>>4,
//   dword j: K=2j+16*half   (pair K,K+1)
__global__ __launch_bounds__(256) void kPrepW(
    const float* __restrict__ Wq_, const float* __restrict__ Wk_,
    const float* __restrict__ Wv_, unsigned* __restrict__ wbuf_)
{
    const int t = blockIdx.x;                 // 0..19
    const int lane = threadIdx.x & 31, kk = threadIdx.x >> 5;
    const float GAS* W;
    int row0;
    if (t < 2)      { W = AS_G(const float, Wq_); row0 = t * 16; }
    else if (t < 4) { W = AS_G(const float, Wk_); row0 = (t - 2) * 16; }
    else            { W = AS_G(const float, Wv_); row0 = (t - 4) * 16; }
    const int m = lane & 15, half = lane >> 4;
    unsigned GAS* dst = AS_G(unsigned, wbuf_) + ((size_t)(t * 8 + kk) * 32 + lane) * 8;
    const float GAS* rp = W + (size_t)(row0 + m) * Cc + kk * 32;
#pragma unroll
    for (int j = 0; j < 8; ++j) {
        int K = 2 * j + 16 * half;
        dst[j] = pk2(rp[K], rp[K + 1]);
    }
}

// B-frag (16-bit B 32x16), same k order as A: lane n=lane&15, dword j: K = 2j + 16*half (pair)
// xt is fp32 LDS tile laid out [c][px], row stride 64
__device__ __forceinline__ v8ui load_x_frag(const float* xt, int px, int k0, int half) {
    v8ui b;
#pragma unroll
    for (int j = 0; j < 8; ++j) {
        int c = k0 + 2 * j + 16 * half;
        b[j] = pk2(xt[c * 64 + px], xt[(c + 1) * 64 + px]);
    }
    return b;
}

// ---------------- Kernel 1: fused QKV projection --------------------
__global__ __launch_bounds__(256) void kProj(
    const float* __restrict__ x,
    const unsigned* __restrict__ wpk_,
    const float* __restrict__ bq_, const float* __restrict__ bk_,
    const float* __restrict__ bv_,
    unsigned short* __restrict__ qb_, unsigned short* __restrict__ kb_,
    unsigned short* __restrict__ vb_)
{
    __shared__ float xt[Cc * 64];              // 64 px x 256 ch fp32 tile (64 KB)
    const int b  = blockIdx.y;
    const int n0 = blockIdx.x * 64;
    const int tid = threadIdx.x;
    const float GAS* xb = AS_G(const float, x) + (size_t)b * Cc * Nn;
    const unsigned GAS* wb = AS_G(const unsigned, wpk_);
    const float GAS* bq = AS_G(const float, bq_);
    const float GAS* bk = AS_G(const float, bk_);
    const float GAS* bv = AS_G(const float, bv_);
    unsigned short GAS* qb = AS_G(unsigned short, qb_);
    unsigned short GAS* kb = AS_G(unsigned short, kb_);
    unsigned short GAS* vb = AS_G(unsigned short, vb_);

#if HAVE_TDM
    // ---- Tensor Data Mover: DMA 2-D tile [256 c][64 px] fp32 global -> LDS ----
    if (tid < 32) {                            // wave-uniform gate (TDM ignores EXEC)
        unsigned lds = (unsigned)(size_t)(void*)&xt[0];
        unsigned long long ga = (unsigned long long)(size_t)(const void*)(x
                              + (size_t)b * Cc * Nn + n0);
        v4u g0;
        g0[0] = 1u;                                            // count=1, user D#
        g0[1] = lds;                                           // lds_addr
        g0[2] = (unsigned)(ga & 0xFFFFFFFFu);                  // global_addr[31:0]
        g0[3] = (unsigned)((ga >> 32) & 0x1FFFFFFu) | (2u << 30); // addr[56:32], type=2
        v8i g1;
        g1[0] = (int)(2u << 16);                               // data_size=4B
        g1[1] = (int)((Nn & 0xFFFFu) << 16);                   // tensor_dim0 lo16
        g1[2] = (int)((Nn >> 16) & 0xFFFFu) | (int)((Cc & 0xFFFFu) << 16);
        g1[3] = (int)((Cc >> 16) & 0xFFFFu) | (int)(64u << 16);// dim1 hi | tile_dim0=64
        g1[4] = (int)(Cc & 0xFFFFu);                           // tile_dim1=256
        g1[5] = (int)Nn;                                       // tensor_dim0_stride lo32
        g1[6] = 0;
        g1[7] = 0;
        v4i gz = {0, 0, 0, 0};
#if defined(__clang_major__) && (__clang_major__ >= 23)
        v8i z8 = {0, 0, 0, 0, 0, 0, 0, 0};
        __builtin_amdgcn_tensor_load_to_lds(g0, g1, gz, gz, z8, 0);
#else
        __builtin_amdgcn_tensor_load_to_lds(g0, g1, gz, gz, 0);
#endif
#if __has_builtin(__builtin_amdgcn_s_wait_tensorcnt)
        __builtin_amdgcn_s_wait_tensorcnt(0);
#else
        asm volatile("s_wait_tensorcnt 0x0" ::: "memory");
#endif
    }
#else
    for (int idx = tid; idx < Cc * 64; idx += 256) {
        int c = idx >> 6, px = idx & 63;
        xt[c * 64 + px] = xb[(size_t)c * Nn + n0 + px];
    }
#endif
    __syncthreads();

    const int wave = tid >> 5, lane = tid & 31;
    const int h = wave >> 2;                // D-tile half
    const int px0 = (wave & 3) * 16;        // pixel sub-tile
    const int m = lane & 15, half = lane >> 4;
    const int pxg = n0 + px0 + m;

    // preload all 8 B-fragments (bf16) for this wave's pixel sub-tile
    v8ui bfr[8];
#pragma unroll
    for (int s = 0; s < 8; ++s) bfr[s] = load_x_frag(xt, px0 + m, s * 32, half);

    // GEMM over one 16-row output tile: pre-packed A-frags, zero conversion VALU
    auto gemm = [&](int t) -> v8f {
        v8f acc = vzero8();
#pragma unroll
        for (int s = 0; s < 8; ++s) {
            v8ui a = *(const v8ui GAS*)(wb + ((size_t)(t * 8 + s) * 32 + lane) * 8);
            acc = wmma_bf16(a, bfr[s], acc);
        }
        return acc;
    };

    if (h == 0) {
        // ---- q: raw logits (softmax over N deferred) ----
        for (int t = 0; t < 2; ++t) {
            v8f acc = gemm(t);
#pragma unroll
            for (int j = 0; j < 8; ++j) {
                int r = t * 16 + j + 8 * half;            // D layout: M = j + 8*half
                qb[(size_t)(b * RC + r) * Nn + pxg] = f2bf(acc[j] + bq[r]);
            }
        }
        // ---- k: both 16-row tiles, softmax over 32 channels per pixel ----
        v8f a0 = gemm(2);
        v8f a1 = gemm(3);
        float e[16];
        float mx = -3.0e38f;
#pragma unroll
        for (int j = 0; j < 8; ++j) {
            float v0 = a0[j] + bk[j + 8 * half];
            float v1 = a1[j] + bk[16 + j + 8 * half];
            e[j] = v0; e[8 + j] = v1;
            mx = fmaxf(mx, fmaxf(v0, v1));
        }
        mx = fmaxf(mx, __shfl_xor(mx, 16));
        float s = 0.f;
#pragma unroll
        for (int i = 0; i < 16; ++i) { e[i] = __expf(e[i] - mx); s += e[i]; }
        s += __shfl_xor(s, 16);
        float inv = 1.f / s;
#pragma unroll
        for (int j = 0; j < 8; ++j) {
            int r0 = j + 8 * half, r1 = 16 + j + 8 * half;
            kb[(size_t)(b * RC + r0) * Nn + pxg] = f2bf(e[j] * inv);
            kb[(size_t)(b * RC + r1) * Nn + pxg] = f2bf(e[8 + j] * inv);
        }
        // ---- v tiles 0..5 ----
        for (int t = 0; t < 6; ++t) {
            v8f acc = gemm(4 + t);
#pragma unroll
            for (int j = 0; j < 8; ++j) {
                int d = t * 16 + j + 8 * half;
                vb[(size_t)(b * Cc + d) * Nn + pxg] = f2bf(acc[j] + bv[d]);
            }
        }
    } else {
        // ---- v tiles 6..15 ----
        for (int t = 6; t < 16; ++t) {
            v8f acc = gemm(4 + t);
#pragma unroll
            for (int j = 0; j < 8; ++j) {
                int d = t * 16 + j + 8 * half;
                vb[(size_t)(b * Cc + d) * Nn + pxg] = f2bf(acc[j] + bv[d]);
            }
        }
    }
}

// ---------------- Kernel 2: q row softmax stats (max, sumexp over N) ----
__global__ __launch_bounds__(256) void kQStats(
    const unsigned short* __restrict__ qb_,
    float* __restrict__ rowmax, float* __restrict__ rowsum)
{
    const int r = blockIdx.x, b = blockIdx.y;
    const unsigned GAS* row32 =
        AS_G(const unsigned, qb_ + (size_t)(b * RC + r) * Nn);
    float m = -3.0e38f, s = 0.f;
    for (int i = threadIdx.x; i < Nn / 2; i += 256) {
        unsigned w = row32[i];
        float v0 = bf2f((unsigned short)(w & 0xFFFFu));
        float v1 = bf2f((unsigned short)(w >> 16));
        float vm = fmaxf(v0, v1);
        if (vm > m) { s *= __expf(m - vm); m = vm; }
        s += __expf(v0 - m) + __expf(v1 - m);
    }
    __shared__ float sm[256], ss[256];
    sm[threadIdx.x] = m; ss[threadIdx.x] = s;
    __syncthreads();
    for (int st = 128; st > 0; st >>= 1) {
        if (threadIdx.x < st) {
            float m2 = sm[threadIdx.x + st], s2 = ss[threadIdx.x + st];
            float mm = fmaxf(sm[threadIdx.x], m2);
            ss[threadIdx.x] = ss[threadIdx.x] * __expf(sm[threadIdx.x] - mm)
                            + s2 * __expf(m2 - mm);
            sm[threadIdx.x] = mm;
        }
        __syncthreads();
    }
    if (threadIdx.x == 0) { rowmax[b * RC + r] = sm[0]; rowsum[b * RC + r] = ss[0]; }
}

// ---------------- Kernel 3: context[b,r,d] = sum_n k'[r,n] * v[d,n] ----
__global__ __launch_bounds__(256) void kContext(
    const unsigned short* __restrict__ kb_,
    const unsigned short* __restrict__ vb_,
    float* __restrict__ ctx)
{
    const int b = blockIdx.y;
    const int nbase = blockIdx.x * 2048;
    const int wave = threadIdx.x >> 5, lane = threadIdx.x & 31;
    const int d0 = wave * 32;              // each wave: 2 d-tiles x 2 r-tiles
    const int m = lane & 15, half = lane >> 4;
    const unsigned short GAS* kbb = AS_G(const unsigned short, kb_) + (size_t)b * RC * Nn;
    const unsigned short GAS* vbb = AS_G(const unsigned short, vb_) + (size_t)b * Cc * Nn;

    v8f acc[2][2];
#pragma unroll
    for (int i = 0; i < 2; ++i)
#pragma unroll
        for (int j = 0; j < 2; ++j) acc[i][j] = vzero8();

    for (int k0 = nbase; k0 < nbase + 2048; k0 += 32) {
        v8ui A[2], Bf[2];
#pragma unroll
        for (int rt = 0; rt < 2; ++rt) {   // A = k' [r x n] (n contiguous, bf16 pairs)
            const unsigned short GAS* rp = kbb + (size_t)(rt * 16 + m) * Nn + k0;
#pragma unroll
            for (int j = 0; j < 8; ++j) {
                int K = 2 * j + 16 * half;
                A[rt][j] = *(const unsigned GAS*)(rp + K);
            }
        }
#pragma unroll
        for (int dt = 0; dt < 2; ++dt) {   // B = v^T [n x d] : row d, n offsets
            const unsigned short GAS* vp = vbb + (size_t)(d0 + dt * 16 + m) * Nn + k0;
#pragma unroll
            for (int j = 0; j < 8; ++j) {
                int K = 2 * j + 16 * half;
                Bf[dt][j] = *(const unsigned GAS*)(vp + K);
            }
        }
#pragma unroll
        for (int rt = 0; rt < 2; ++rt)
#pragma unroll
            for (int dt = 0; dt < 2; ++dt)
                acc[rt][dt] = wmma_bf16(A[rt], Bf[dt], acc[rt][dt]);
    }
#pragma unroll
    for (int rt = 0; rt < 2; ++rt)
#pragma unroll
        for (int dt = 0; dt < 2; ++dt) {
            int d = d0 + dt * 16 + (lane & 15);          // D: N = lane&15
#pragma unroll
            for (int j = 0; j < 8; ++j) {
                int r = rt * 16 + j + 8 * half;          // D: M = j + 8*half
                atomicAdd(&ctx[(size_t)(b * RC + r) * Cc + d], acc[rt][dt][j]);
            }
        }
}

// ---------------- Kernel 4: out = gamma * (softmax(q)^T ctx') + x ----
__global__ __launch_bounds__(256) void kOut(
    const unsigned short* __restrict__ qb_,
    const float* __restrict__ ctx_,
    const float* __restrict__ rowmax_, const float* __restrict__ rowsum_,
    const float* __restrict__ x_, const float* __restrict__ gamma_,
    float* __restrict__ out_)
{
    __shared__ float tile[16 * 129];
    const int b = blockIdx.y;
    const int n0 = blockIdx.x * 128;
    const int wave = threadIdx.x >> 5, lane = threadIdx.x & 31;
    const int px0 = wave * 16;
    const int m = lane & 15, half = lane >> 4;
    const int pxg = n0 + px0 + m;
    const unsigned short GAS* qbb = AS_G(const unsigned short, qb_) + (size_t)b * RC * Nn;
    const float GAS* ctx = AS_G(const float, ctx_);
    const float GAS* rowmax = AS_G(const float, rowmax_);
    const float GAS* rowsum = AS_G(const float, rowsum_);
    const float GAS* x = AS_G(const float, x_);
    float GAS* out = AS_G(float, out_);
    const float g = AS_G(const float, gamma_)[0];

    // A-frag: p^T [px x r], p = exp(q - rowmax[r]); reused for all 16 d-tiles
    v8ui A;
#pragma unroll
    for (int j = 0; j < 8; ++j) {
        int r = 2 * j + 16 * half;
        float q0 = bf2f(qbb[(size_t)r * Nn + pxg]);
        float q1 = bf2f(qbb[(size_t)(r + 1) * Nn + pxg]);
        A[j] = pk2(__expf(q0 - rowmax[b * RC + r]),
                   __expf(q1 - rowmax[b * RC + r + 1]));
    }

    for (int t = 0; t < 16; ++t) {
        const int d0 = t * 16;
        v8ui Bf;                            // ctx'[r x d], 1/Z folded in
#pragma unroll
        for (int j = 0; j < 8; ++j) {
            int r = 2 * j + 16 * half;
            float z0 = 1.f / rowsum[b * RC + r];
            float z1 = 1.f / rowsum[b * RC + r + 1];
            Bf[j] = pk2(ctx[(size_t)(b * RC + r) * Cc + d0 + m] * z0,
                        ctx[(size_t)(b * RC + r + 1) * Cc + d0 + m] * z1);
        }
        v8f acc = wmma_bf16(A, Bf, vzero8());
        // D tile [px x d] -> LDS transposed [d][px] for coalesced [C,N] stores
#pragma unroll
        for (int j = 0; j < 8; ++j)
            tile[m * 129 + px0 + j + 8 * half] = acc[j];
        __syncthreads();
        for (int idx = threadIdx.x; idx < 2048; idx += 256) {
            int row = idx >> 7, col = idx & 127;
            size_t off = (size_t)(b * Cc + d0 + row) * Nn + n0 + col;
            out[off] = g * tile[row * 129 + col] + x[off];
        }
        __syncthreads();
    }
}

// ---------------- host launcher ----------------
extern "C" void kernel_launch(void* const* d_in, const int* in_sizes, int n_in,
                              void* d_out, int out_size, void* d_ws, size_t ws_size,
                              hipStream_t stream) {
    const float* x     = (const float*)d_in[0];
    const float* Wq    = (const float*)d_in[1];
    const float* bq    = (const float*)d_in[2];
    const float* Wk    = (const float*)d_in[3];
    const float* bk    = (const float*)d_in[4];
    const float* Wv    = (const float*)d_in[5];
    const float* bv    = (const float*)d_in[6];
    const float* gamma = (const float*)d_in[7];
    float* out = (float*)d_out;

    // workspace layout (~168 MB)
    char* ws = (char*)d_ws;
    size_t off = 0;
    unsigned short* qb = (unsigned short*)(ws + off); off += (size_t)Bb * RC * Nn * 2;  // 16 MB
    unsigned short* kb = (unsigned short*)(ws + off); off += (size_t)Bb * RC * Nn * 2;  // 16 MB
    unsigned short* vb = (unsigned short*)(ws + off); off += (size_t)Bb * Cc * Nn * 2;  // 134 MB
    float* ctx    = (float*)(ws + off); off += (size_t)Bb * RC * Cc * 4;                // 128 KB
    float* rowmax = (float*)(ws + off); off += (size_t)Bb * RC * 4;
    float* rowsum = (float*)(ws + off); off += (size_t)Bb * RC * 4;
    unsigned* wpk = (unsigned*)(ws + off); off += (size_t)20 * 8 * 32 * 8 * 4;          // 160 KB

    kZero<<<(Bb * RC * Cc + 255) / 256, 256, 0, stream>>>(ctx, Bb * RC * Cc);
    kPrepW<<<dim3(20), 256, 0, stream>>>(Wq, Wk, Wv, wpk);
    kProj<<<dim3(Nn / 64, Bb), 256, 0, stream>>>(x, wpk, bq, bk, bv, qb, kb, vb);
    kQStats<<<dim3(RC, Bb), 256, 0, stream>>>(qb, rowmax, rowsum);
    kContext<<<dim3(Nn / 2048, Bb), 256, 0, stream>>>(kb, vb, ctx);
    kOut<<<dim3(Nn / 128, Bb), 256, 0, stream>>>(qb, ctx, rowmax, rowsum, x, gamma, out);
}
